// LSTM_30537217475003
// MI455X (gfx1250) — compile-verified, matrix operand order checked by reading, not run on hardware
//
#include <hip/hip_runtime.h>

typedef __attribute__((ext_vector_type(16))) __bf16 v16bf;
typedef __attribute__((ext_vector_type(8)))  __bf16 v8bf;
typedef __attribute__((ext_vector_type(8)))  float  v8f;
typedef __attribute__((ext_vector_type(4)))  unsigned int v4u;
typedef __attribute__((ext_vector_type(8)))  int    v8i;
typedef __attribute__((ext_vector_type(4)))  int    v4i;

#define B_   64
#define T_   512
#define I_   512
#define H_   512
#define G4_  2048   // 4*H
#define KTOT 1024   // I + H
#define KSTG 512    // K columns per LDS stage (2 stages/step, double-buffered)

#if defined(__has_builtin)
#if __has_builtin(__builtin_amdgcn_tensor_load_to_lds)
#define HAS_TDM 1
#endif
#endif

// ---- workspace layout (bytes) ----
#define XB_OFF  0u            // bf16 x    [B,T,I]   : 33554432
#define WC_OFF  33554432u     // bf16 wcat [4H,I+H]  :  4194304
#define C_OFF   37748736u     // f32  c    [B,H]     :   131072
#define H0_OFF  37879808u     // bf16 h0   [B,H]     :    65536
#define H1_OFF  37945344u     // bf16 h1   [B,H]     :    65536
#define BS_OFF  38010880u     // f32  bsum [4H]      :     8192

__device__ __forceinline__ v16bf load_frag16(const __bf16* p) {
    union { v16bf v; v8bf h[2]; } u;
    u.h[0] = *reinterpret_cast<const v8bf*>(p);       // K .. K+7
    u.h[1] = *reinterpret_cast<const v8bf*>(p + 16);  // K+16 .. K+23
    return u.v;
}

__device__ __forceinline__ float sigm(float x) {
    return 1.0f / (1.0f + __expf(-x));
}

// The TDM descriptor refers to LDS only by integer address, so the compiler
// cannot see the LDS write. Escape the LDS pointer into an asm with a memory
// clobber so subsequent ds_loads of the staged weights are not folded to undef.
__device__ __forceinline__ void lds_publish(const void* p) {
    asm volatile("" : : "v"(p) : "memory");
}

// Issue one TDM 2D tile load: 16 rows x KSTG cols of bf16 from wcat into LDS.
__device__ __forceinline__ void tdm_load_tile(unsigned long long gaddr,
                                              unsigned int lds_off) {
#ifdef HAS_TDM
    v4u g0;
    g0.x = 1u;                                         // count=1, user mode
    g0.y = lds_off;                                    // lds_addr (static LDS base == 0)
    g0.z = (unsigned int)(gaddr & 0xffffffffull);      // global_addr[31:0]
    g0.w = (unsigned int)((gaddr >> 32) & 0x1ffffffull)
         | (2u << 30);                                 // global_addr[56:32] | type=2
    v8i g1;
    g1[0] = (int)(1u << 16);                           // data_size=1 (2 bytes), mask=0
    g1[1] = (int)(((unsigned)KSTG & 0xffffu) << 16);   // tensor_dim0 = 512 (lo16)
    g1[2] = (int)((16u & 0xffffu) << 16);              // dim0 hi=0 | tensor_dim1 = 16
    g1[3] = (int)(((unsigned)KSTG & 0xffffu) << 16);   // dim1 hi=0 | tile_dim0 = 512
    g1[4] = 16;                                        // tile_dim1=16 | tile_dim2=0
    g1[5] = KTOT;                                      // tensor_dim0_stride = 1024 (lo32)
    g1[6] = 0;
    g1[7] = 0;
    v4i z4 = {0, 0, 0, 0};
#if __clang_major__ >= 23
    v8i z8 = {0, 0, 0, 0, 0, 0, 0, 0};
    __builtin_amdgcn_tensor_load_to_lds(g0, g1, z4, z4, z8, 0);
#else
    __builtin_amdgcn_tensor_load_to_lds(g0, g1, z4, z4, 0);
#endif
#endif
}

// One LSTM timestep. Grid: 32 blocks x 128 threads (4 waves).
// block -> htile in [0,32); wave -> mtile in [0,4).
// Weights for the block's 64 gate-rows are staged to LDS by the TDM into two
// 64KB buffers; both DMA stages are issued up front so the w_hh stage streams
// in while the x-part WMMAs run.
__global__ void __launch_bounds__(128)
lstm_step(const __bf16* __restrict__ xb,    // [B,T,I] bf16
          const __bf16* __restrict__ wcat,  // [4H, I+H] bf16 row-major
          const float*  __restrict__ bsum,  // [4H] b_ih+b_hh
          const __bf16* __restrict__ hprev, // [B,H] bf16
          __bf16*       __restrict__ hnext, // [B,H] bf16
          float*        __restrict__ c,     // [B,H] f32 (in-place)
          float*        __restrict__ out,   // [T,B,H] f32
          int t)
{
    __shared__ __bf16 wlds[2][4][16][KSTG];   // 128 KB, assumed at LDS offset 0

    const int tid   = threadIdx.x;
    const int wave  = tid >> 5;             // 0..3  -> mtile
    const int lane  = tid & 31;
    const int htile = blockIdx.x;           // 0..31
    const int h0    = htile * 16;
    const int r     = lane & 15;
    const int halfk = lane >> 4;

    const __bf16* abaseX = xb    + (size_t)(wave * 16 + r) * (T_ * I_) + (size_t)t * I_;
    const __bf16* abaseH = hprev + (size_t)(wave * 16 + r) * H_;

    v8f acc[4] = {};

#ifdef HAS_TDM
    if (wave == 0) {
        // stage 0: cols [0,512) -> buf0 ; stage 1: cols [512,1024) -> buf1
        #pragma unroll
        for (int g = 0; g < 4; ++g) {
            unsigned long long ga = (unsigned long long)(uintptr_t)
                (wcat + ((size_t)g * 512 + h0) * KTOT);
            tdm_load_tile(ga, (unsigned int)(g * 16 * KSTG * 2));
        }
        #pragma unroll
        for (int g = 0; g < 4; ++g) {
            unsigned long long ga = (unsigned long long)(uintptr_t)
                (wcat + ((size_t)g * 512 + h0) * KTOT + KSTG);
            tdm_load_tile(ga, (unsigned int)(65536 + g * 16 * KSTG * 2));
        }
        __builtin_amdgcn_s_wait_tensorcnt(4);   // in-order: first 4 (stage 0) done
    }
    __syncthreads();
    lds_publish(&wlds[0][0][0][0]);
#else
    // Fallback: cooperative copy of both stages up front.
    for (int i = tid; i < 2 * 4 * 16 * (KSTG / 8); i += blockDim.x) {
        int s   = i / (4 * 16 * (KSTG / 8));
        int rem = i % (4 * 16 * (KSTG / 8));
        int row = rem / (KSTG / 8);         // 0..63
        int seg = rem % (KSTG / 8);         // 8 bf16 per seg
        int g = row >> 4, rr = row & 15;
        *(v8bf*)&wlds[s][g][rr][seg * 8] =
            *(const v8bf*)(wcat + ((size_t)g * 512 + h0 + rr) * KTOT + s * KSTG + seg * 8);
    }
    __syncthreads();
#endif

    // K = 0..511 : x_t @ w_ih^T, B-fragments from LDS buf0 (w_hh DMA in flight)
    #pragma unroll 4
    for (int kt = 0; kt < 16; ++kt) {
        const int ko = kt * 32 + halfk * 8;
        v16bf a = load_frag16(abaseX + ko);
        #pragma unroll
        for (int g = 0; g < 4; ++g) {
            v16bf bb = load_frag16(&wlds[0][g][r][ko]);
            acc[g] = __builtin_amdgcn_wmma_f32_16x16x32_bf16(
                false, a, false, bb, (short)0, acc[g], false, false);
        }
    }

#ifdef HAS_TDM
    if (wave == 0)
        __builtin_amdgcn_s_wait_tensorcnt(0);   // stage 1 (w_hh) landed
    __syncthreads();
    lds_publish(&wlds[1][0][0][0]);
#endif

    // K = 512..1023 : h_{t-1} @ w_hh^T, B-fragments from LDS buf1
    #pragma unroll 4
    for (int kt = 0; kt < 16; ++kt) {
        const int ko = kt * 32 + halfk * 8;
        v16bf a = load_frag16(abaseH + ko);
        #pragma unroll
        for (int g = 0; g < 4; ++g) {
            v16bf bb = load_frag16(&wlds[1][g][r][ko]);
            acc[g] = __builtin_amdgcn_wmma_f32_16x16x32_bf16(
                false, a, false, bb, (short)0, acc[g], false, false);
        }
    }

    // Fused LSTM cell. C/D layout: lane holds column n = lane&15;
    // VGPR v holds row m = v + 8*(lane>>4).
    const int n = lane & 15;
    float bs[4];
    #pragma unroll
    for (int g = 0; g < 4; ++g)
        bs[g] = bsum[g * H_ + h0 + n];

    #pragma unroll
    for (int v = 0; v < 8; ++v) {
        const int brow = wave * 16 + v + halfk * 8;
        const float gi = acc[0][v] + bs[0];
        const float gf = acc[1][v] + bs[1];
        const float gg = acc[2][v] + bs[2];
        const float go = acc[3][v] + bs[3];

        const float si = sigm(gi);
        const float sf = sigm(gf);
        const float tg = tanhf(gg);
        const float so = sigm(go);

        const size_t ci = (size_t)brow * H_ + h0 + n;
        const float cn = sf * c[ci] + si * tg;
        c[ci] = cn;
        const float hn = so * tanhf(cn);
        out[(size_t)t * (B_ * H_) + ci] = hn;   // out[t][b][h]
        hnext[ci] = (__bf16)hn;
    }
}

__global__ void k_f32_to_bf16(const float* __restrict__ src, __bf16* __restrict__ dst, size_t n) {
    size_t i = (size_t)blockIdx.x * blockDim.x + threadIdx.x;
    if (i < n) dst[i] = (__bf16)src[i];
}

// Pack [w_ih | w_hh] -> wcat[4H][I+H] bf16, row-major.
__global__ void k_pack_w(const float* __restrict__ w_ih, const float* __restrict__ w_hh,
                         __bf16* __restrict__ wcat) {
    size_t i = (size_t)blockIdx.x * blockDim.x + threadIdx.x;
    if (i >= (size_t)G4_ * KTOT) return;
    const size_t row = i / KTOT;
    const size_t col = i % KTOT;
    const float v = (col < I_) ? w_ih[row * I_ + col]
                               : w_hh[row * H_ + (col - I_)];
    wcat[i] = (__bf16)v;
}

__global__ void k_init(float* __restrict__ c, __bf16* __restrict__ h0, __bf16* __restrict__ h1,
                       const float* __restrict__ b_ih, const float* __restrict__ b_hh,
                       float* __restrict__ bsum) {
    int i = blockIdx.x * blockDim.x + threadIdx.x;
    if (i < B_ * H_) {
        c[i]  = 0.0f;
        h0[i] = (__bf16)0.0f;
        h1[i] = (__bf16)0.0f;
    }
    if (i < G4_) bsum[i] = b_ih[i] + b_hh[i];
}

extern "C" void kernel_launch(void* const* d_in, const int* in_sizes, int n_in,
                              void* d_out, int out_size, void* d_ws, size_t ws_size,
                              hipStream_t stream) {
    const float* x    = (const float*)d_in[0];
    const float* w_ih = (const float*)d_in[1];
    const float* w_hh = (const float*)d_in[2];
    const float* b_ih = (const float*)d_in[3];
    const float* b_hh = (const float*)d_in[4];
    float* out = (float*)d_out;

    char* ws = (char*)d_ws;
    __bf16* xb   = (__bf16*)(ws + XB_OFF);
    __bf16* wcat = (__bf16*)(ws + WC_OFF);
    float*  cst  = (float*) (ws + C_OFF);
    __bf16* hb0  = (__bf16*)(ws + H0_OFF);
    __bf16* hb1  = (__bf16*)(ws + H1_OFF);
    float*  bsum = (float*) (ws + BS_OFF);

    const size_t nx = (size_t)B_ * T_ * I_;          // 16,777,216
    const size_t nw = (size_t)G4_ * KTOT;            //  2,097,152

    k_f32_to_bf16<<<dim3((unsigned)((nx + 255) / 256)), dim3(256), 0, stream>>>(x, xb, nx);
    k_pack_w<<<dim3((unsigned)((nw + 255) / 256)), dim3(256), 0, stream>>>(w_ih, w_hh, wcat);
    k_init<<<dim3((B_ * H_ + 255) / 256), dim3(256), 0, stream>>>(cst, hb0, hb1, b_ih, b_hh, bsum);

    for (int t = 0; t < T_; ++t) {
        const __bf16* hp = (t & 1) ? hb1 : hb0;
        __bf16*       hn = (t & 1) ? hb0 : hb1;
        lstm_step<<<dim3(32), dim3(128), 0, stream>>>(xb, wcat, bsum, hp, hn, cst, out, t);
    }
}
